// MultiheadAttention_61718680044069
// MI455X (gfx1250) — matrix operand order repaired and checked by reading, not compiled, not run on hardware
//
#include <hip/hip_runtime.h>
#include <hip/hip_bf16.h>
#include <stdint.h>

// Problem constants (match reference)
#define E   1024
#define NH  16
#define HD  64
#define TT  1024
#define BB  4
#define SS  1024

typedef __attribute__((ext_vector_type(16))) _Float16 v16h;
typedef __attribute__((ext_vector_type(8)))  float    v8f;

// ---------------- CDNA5 TDM (Tensor Data Mover) support -------------------
#if defined(__gfx1250__) && __has_builtin(__builtin_amdgcn_tensor_load_to_lds) && \
    __has_builtin(__builtin_amdgcn_s_wait_tensorcnt)
#define HAS_TDM 1
#if __has_include(<hip/amd_detail/amd_gfx1250_TDM.h>)
#define TDM_ARGS 6   // amdgpu-toolchain / therock headers -> 6-arg builtin
#else
#define TDM_ARGS 5   // ROCm 7.2 clang-22 -> 5-arg builtin
#endif
#else
#define HAS_TDM 0
#endif

#if HAS_TDM
typedef __attribute__((ext_vector_type(4))) unsigned int u32x4;
typedef __attribute__((ext_vector_type(8))) int          i32x8;
typedef __attribute__((ext_vector_type(4))) int          i32x4;

__device__ __forceinline__ uint32_t lds_byte_off(const void* p) {
  // LDS aperture: low 32 bits of the flat address are the LDS byte offset.
  return (uint32_t)(uintptr_t)p;
}

// 2D tile load Global->LDS via TDM. Element size = 2 bytes (f16).
// Tile rows are tile_d0 elements; LDS rows padded by 16B (pad codes 3/3:
// pad after every 16 DWORDs (=64B row), pad amount 4 DWORDs (=16B)).
__device__ __forceinline__ void tdm_load_tile_f16(
    const void* gsrc, uint32_t lds_byte, uint64_t row_stride_elems,
    uint32_t tile_d0, uint32_t tile_d1)
{
  uint64_t ga = (uint64_t)(uintptr_t)gsrc;
  u32x4 g0;
  g0.x = 1u;                                               // count=1 (valid D#)
  g0.y = lds_byte;                                         // lds_addr
  g0.z = (uint32_t)ga;                                     // global_addr[31:0]
  g0.w = (uint32_t)((ga >> 32) & 0x1FFFFFFu) | (2u << 30); // addr[56:32]|type=2
  i32x8 g1;
  g1[0] = (1 << 16)        // data_size = 1 -> 2 bytes
        | (1 << 20)        // pad_enable
        | (3 << 22)        // pad_interval: 16 DWORDs (64B) between pads
        | (3 << 25);       // pad_amount: 4 DWORDs (16B)
  g1[1] = (int)((tile_d0 & 0xFFFFu) << 16);                // tensor_dim0[15:0]
  g1[2] = (int)((tile_d0 >> 16) | ((tile_d1 & 0xFFFFu) << 16)); // dim0 hi | dim1 lo
  g1[3] = (int)((tile_d1 >> 16) | (tile_d0 << 16));        // dim1 hi | tile_dim0
  g1[4] = (int)tile_d1;                                    // tile_dim1 | tile_dim2=0
  g1[5] = (int)(uint32_t)row_stride_elems;                 // dim0_stride[31:0]
  g1[6] = (int)(uint32_t)(row_stride_elems >> 32);         // dim0_stride[47:32]
  g1[7] = 0;
  i32x4 z4 = {0, 0, 0, 0};
#if TDM_ARGS == 6
  i32x8 z8 = {0, 0, 0, 0, 0, 0, 0, 0};
  __builtin_amdgcn_tensor_load_to_lds(g0, g1, z4, z4, z8, 0);
#else
  __builtin_amdgcn_tensor_load_to_lds(g0, g1, z4, z4, 0);
#endif
}
#endif  // HAS_TDM

__device__ __forceinline__ void tensor_wait0() {
#if HAS_TDM
  __builtin_amdgcn_s_wait_tensorcnt(0);
#endif
}

// ---------------- WMMA fragment helpers -----------------------------------
union Frag { v16h v; uint32_t u[8]; };

// A fragment: 16x32 f16, source row-major MxK, base at (row0,k0).
// Lanes 0-15 rows; VGPR v holds K pair = (v>=4?16:0) + 8*(lane>=16) + 2*(v&3)
__device__ __forceinline__ v16h frag_a(const _Float16* base, int ld, int lane) {
  int m  = lane & 15;
  int kb = (lane >> 4) << 3;
  Frag f;
  const _Float16* p = base + (size_t)m * ld + kb;
#pragma unroll
  for (int v = 0; v < 8; ++v) {
    int koff = ((v & 4) << 2) + ((v & 3) << 1);
    f.u[v] = *(const uint32_t*)(p + koff);
  }
  return f.v;
}

// B fragment: 32x16 (KxN) f16, source stored N-major [N][K], base at (n0,k0).
// Lanes 0-15 cols; VGPR v holds K pair = (v>=4?16:0) + 8*(lane>=16) + 2*(v&3)
__device__ __forceinline__ v16h frag_b(const _Float16* base, int ld, int lane) {
  int n  = lane & 15;
  int kb = (lane >> 4) << 3;
  Frag f;
  const _Float16* p = base + (size_t)n * ld + kb;
#pragma unroll
  for (int v = 0; v < 8; ++v) f.u[v] = *(const uint32_t*)(p + ((v & 4) << 2) + ((v & 3) << 1));
  return f.v;
}

__device__ __forceinline__ v8f wmma_f16(v16h a, v16h b, v8f c) {
  return __builtin_amdgcn_wmma_f32_16x16x32_f16(false, a, false, b, (short)0, c,
                                                false, false);
}

enum { MODE_Q = 0, MODE_K = 1, MODE_VT = 2, MODE_ATTNOUT = 3, MODE_FINAL = 4 };

#define BM  128
#define BN  64
#define BK  32
#define BKP 40   // padded LDS K-stride (80B rows: conflict-free + 16B aligned)

// Generic WMMA GEMM: C[M,N] = A[M,K] * B_NMajor[N,K]^T, bias/scale epilogue,
// compile-time mode. 256 threads = 8 waves, wave tile 32x32, double-buffered
// LDS staged by TDM (f16 tiles) or vectorized loads (fp32 A converts inline).
template <int MODE, bool AHALF>
__global__ __launch_bounds__(256) void gemm_kernel(
    const void* __restrict__ Aptr, int lda,
    const _Float16* __restrict__ Bn, int ldb,
    const float* __restrict__ bias, float scale,
    int M, int N, int K,
    void* __restrict__ outp)
{
  __shared__ _Float16 As[2][BM][BKP];
  __shared__ _Float16 Bs[2][BN][BKP];
  int tid = threadIdx.x, lane = tid & 31, wave = tid >> 5;
  int m0 = blockIdx.x * BM, n0 = blockIdx.y * BN;
  int bh = blockIdx.z;
  const _Float16* Ah = (const _Float16*)Aptr;
  const float*    Af = (const float*)Aptr;
  if (MODE == MODE_ATTNOUT) {           // per-(b,h) sub-problem
    Ah += (size_t)bh * TT * SS;         // attn weights [T,S]
    Bn += (size_t)bh * HD * SS;         // v transposed [D,S]
  }
  int wm = (wave >> 1) * 32, wn = (wave & 1) * 32;
  v8f acc00 = {}, acc01 = {}, acc10 = {}, acc11 = {};

  auto stage = [&](int buf, int k0) {
#if HAS_TDM
    if (tid == 0) {   // EXEC ignored by tensor ops; one wave issues the DMA
      tdm_load_tile_f16(Bn + (size_t)n0 * ldb + k0,
                        lds_byte_off(&Bs[buf][0][0]), (uint64_t)ldb, BK, BN);
      if (AHALF)
        tdm_load_tile_f16(Ah + (size_t)m0 * lda + k0,
                          lds_byte_off(&As[buf][0][0]), (uint64_t)lda, BK, BM);
    }
#else
    {   // B tile 64x32 f16: one 16B chunk per thread
      int row = tid >> 2, ch = (tid & 3) << 3;
      *(uint4*)&Bs[buf][row][ch] =
          *(const uint4*)(Bn + (size_t)(n0 + row) * ldb + k0 + ch);
    }
    if (AHALF) {       // A tile 128x32 f16: two 16B chunks per thread
#pragma unroll
      for (int i = 0; i < 2; ++i) {
        int p = tid + i * 256;
        int row = p >> 2, ch = (p & 3) << 3;
        *(uint4*)&As[buf][row][ch] =
            *(const uint4*)(Ah + (size_t)(m0 + row) * lda + k0 + ch);
      }
    }
#endif
    if (!AHALF) {      // fp32 A tile: load float2, convert to packed f16
#pragma unroll
      for (int i = 0; i < 8; ++i) {
        int p = tid + i * 256;
        int r = p >> 4, c = (p & 15) << 1;
        const float* s = Af + (size_t)(m0 + r) * lda + k0 + c;
        float2 x = *(const float2*)s;
        As[buf][r][c]     = (_Float16)x.x;
        As[buf][r][c + 1] = (_Float16)x.y;
      }
    }
  };

  int nsteps = K / BK;
  stage(0, 0);
  for (int s = 0; s < nsteps; ++s) {
    int buf = s & 1;
    tensor_wait0();        // wave-local: waits this wave's TDM issues
    __syncthreads();       // tile `buf` visible to all waves
    if (s + 1 < nsteps) stage(buf ^ 1, (s + 1) * BK);   // prefetch next tile
    v16h a0 = frag_a(&As[buf][wm][0],      BKP, lane);
    v16h a1 = frag_a(&As[buf][wm + 16][0], BKP, lane);
    v16h b0 = frag_b(&Bs[buf][wn][0],      BKP, lane);
    v16h b1 = frag_b(&Bs[buf][wn + 16][0], BKP, lane);
    acc00 = wmma_f16(a0, b0, acc00);
    acc01 = wmma_f16(a0, b1, acc01);
    acc10 = wmma_f16(a1, b0, acc10);
    acc11 = wmma_f16(a1, b1, acc11);
  }

  int rowoff = (lane >> 4) << 3;
  int nlane  = lane & 15;
#pragma unroll
  for (int ij = 0; ij < 4; ++ij) {
    int i = ij >> 1, j = ij & 1;
    v8f acc = (ij == 0) ? acc00 : (ij == 1) ? acc01 : (ij == 2) ? acc10 : acc11;
#pragma unroll
    for (int r = 0; r < 8; ++r) {
      int m = m0 + wm + i * 16 + r + rowoff;   // C layout: M = r + 8*(lane>=16)
      int n = n0 + wn + j * 16 + nlane;        //           N = lane&15
      float val = acc[r];
      if (MODE != MODE_ATTNOUT) val += bias[n];
      val *= scale;
      if (MODE == MODE_Q || MODE == MODE_K) {
        int t = m >> 2, b = m & 3;             // [T,B,*] row = t*B+b (B=4)
        int h = n >> 6, d = n & 63;
        ((_Float16*)outp)[(((size_t)(b * NH + h) * TT) + t) * HD + d] =
            (_Float16)val;
      } else if (MODE == MODE_VT) {
        int sdx = m >> 2, b = m & 3;
        int h = n >> 6, d = n & 63;
        ((_Float16*)outp)[((size_t)(b * NH + h) * HD + d) * SS + sdx] =
            (_Float16)val;
      } else if (MODE == MODE_ATTNOUT) {
        int b = bh >> 4, h = bh & 15;          // bh = b*H + h
        ((_Float16*)outp)[((size_t)m * BB + b) * E + h * HD + n] = (_Float16)val;
      } else {                                 // MODE_FINAL: fp32 [T,B,E]
        ((float*)outp)[(size_t)m * N + n] = val;
      }
    }
  }
}

// Fused scores + masks + softmax: one block per (b,h,16-row strip of T).
// 8 waves x 128 columns = S=1024. Writes f16 attention weights [B,H,T,S].
__global__ __launch_bounds__(256) void scores_softmax_kernel(
    const _Float16* __restrict__ qbuf, const _Float16* __restrict__ kbuf,
    const float* __restrict__ attn_mask, const unsigned char* __restrict__ pad,
    _Float16* __restrict__ wbuf)
{
  int tid = threadIdx.x, lane = tid & 31, wave = tid >> 5;
  int blk = blockIdx.x;
  int t16 = blk & 63;            // T/16 = 64 strips
  int bh  = blk >> 6;            // b*H + h
  int b   = bh >> 4;
  int t0  = t16 << 4;

  const _Float16* qb = qbuf + ((size_t)bh * TT + t0) * HD;
  v16h a0 = frag_a(qb,      HD, lane);   // K = 0..31 of D
  v16h a1 = frag_a(qb + 32, HD, lane);   // K = 32..63
  int scol = wave * 128;

  v8f acc[8];
#pragma unroll
  for (int jt = 0; jt < 8; ++jt) {
    const _Float16* kb = kbuf + ((size_t)bh * SS + scol + jt * 16) * HD;
    v16h b0 = frag_b(kb,      HD, lane);
    v16h b1 = frag_b(kb + 32, HD, lane);
    v8f c = {};
    c = wmma_f16(a0, b0, c);
    c = wmma_f16(a1, b1, c);
    acc[jt] = c;
  }

  int rowoff = (lane >> 4) << 3;
  int nlane  = lane & 15;
#pragma unroll
  for (int jt = 0; jt < 8; ++jt) {
    int s = scol + jt * 16 + nlane;
    float padd = pad[b * SS + s] ? -1e30f : 0.0f;
#pragma unroll
    for (int r = 0; r < 8; ++r) {
      int t = t0 + r + rowoff;
      acc[jt][r] += attn_mask[(size_t)t * SS + s] + padd;
    }
  }

  __shared__ float red[8][16];
  float rmax[8], rsum[8];
#pragma unroll
  for (int r = 0; r < 8; ++r) {
    float v = acc[0][r];
#pragma unroll
    for (int jt = 1; jt < 8; ++jt) v = fmaxf(v, acc[jt][r]);
#pragma unroll
    for (int off = 1; off < 16; off <<= 1) v = fmaxf(v, __shfl_xor(v, off, 32));
    rmax[r] = v;
  }
  if (lane == 0 || lane == 16) {
#pragma unroll
    for (int r = 0; r < 8; ++r) red[wave][rowoff + r] = rmax[r];
  }
  __syncthreads();
#pragma unroll
  for (int r = 0; r < 8; ++r) {
    float v = red[0][rowoff + r];
#pragma unroll
    for (int w = 1; w < 8; ++w) v = fmaxf(v, red[w][rowoff + r]);
    rmax[r] = v;
  }
  __syncthreads();
#pragma unroll
  for (int r = 0; r < 8; ++r) {
    float s = 0.f;
#pragma unroll
    for (int jt = 0; jt < 8; ++jt) {
      float p = __expf(acc[jt][r] - rmax[r]);
      acc[jt][r] = p;
      s += p;
    }
#pragma unroll
    for (int off = 1; off < 16; off <<= 1) s += __shfl_xor(s, off, 32);
    rsum[r] = s;
  }
  if (lane == 0 || lane == 16) {
#pragma unroll
    for (int r = 0; r < 8; ++r) red[wave][rowoff + r] = rsum[r];
  }
  __syncthreads();
#pragma unroll
  for (int r = 0; r < 8; ++r) {
    float s = 0.f;
#pragma unroll
    for (int w = 0; w < 8; ++w) s += red[w][rowoff + r];
    rsum[r] = 1.0f / s;
  }
#pragma unroll
  for (int jt = 0; jt < 8; ++jt) {
    int s = scol + jt * 16 + nlane;
#pragma unroll
    for (int r = 0; r < 8; ++r) {
      int t = t0 + r + rowoff;
      wbuf[((size_t)bh * TT + t) * SS + s] = (_Float16)(acc[jt][r] * rsum[r]);
    }
  }
}

// avg_weights[b,t,s] = mean over heads of attn weights
__global__ void avg_kernel(const _Float16* __restrict__ wbuf,
                           float* __restrict__ avg) {
  size_t idx = (size_t)blockIdx.x * blockDim.x + threadIdx.x;
  if (idx >= (size_t)BB * TT * SS) return;
  int s = (int)(idx & (SS - 1));
  int t = (int)((idx >> 10) & (TT - 1));
  int b = (int)(idx >> 20);
  float acc = 0.f;
#pragma unroll
  for (int h = 0; h < NH; ++h)
    acc += (float)wbuf[(((size_t)(b * NH + h) * TT) + t) * SS + s];
  avg[idx] = acc * (1.0f / NH);
}

// fp32 -> f16 weight conversion (in_proj_weight and out_w)
__global__ void convert_kernel(const float* __restrict__ w,
                               const float* __restrict__ ow,
                               _Float16* __restrict__ wh,
                               _Float16* __restrict__ owh) {
  size_t idx = (size_t)blockIdx.x * blockDim.x + threadIdx.x;
  const size_t NW = (size_t)3 * E * E;
  if (idx < NW) wh[idx] = (_Float16)w[idx];
  if (idx < (size_t)E * E) owh[idx] = (_Float16)ow[idx];
}

extern "C" void kernel_launch(void* const* d_in, const int* in_sizes, int n_in,
                              void* d_out, int out_size, void* d_ws, size_t ws_size,
                              hipStream_t stream) {
  const float* query = (const float*)d_in[0];
  const float* key   = (const float*)d_in[1];
  const float* value = (const float*)d_in[2];
  const unsigned char* pad = (const unsigned char*)d_in[3];  // bool mask [B,S]
  const float* attn_mask = (const float*)d_in[4];
  const float* ipw  = (const float*)d_in[5];
  const float* ipb  = (const float*)d_in[6];
  const float* outw = (const float*)d_in[7];
  const float* outb = (const float*)d_in[8];

  char* ws = (char*)d_ws;
  _Float16* Wh   = (_Float16*)ws; ws += (size_t)3 * E * E * 2;      // in_proj f16
  _Float16* Owh  = (_Float16*)ws; ws += (size_t)E * E * 2;          // out_w f16
  _Float16* qbuf = (_Float16*)ws; ws += (size_t)TT * BB * E * 2;    // [B,H,T,D]
  _Float16* kbuf = (_Float16*)ws; ws += (size_t)SS * BB * E * 2;    // [B,H,S,D]
  _Float16* vt   = (_Float16*)ws; ws += (size_t)SS * BB * E * 2;    // [B,H,D,S]
  _Float16* aout = (_Float16*)ws; ws += (size_t)TT * BB * E * 2;    // [T,B,E]
  _Float16* wbuf = (_Float16*)ws;                                    // [B,H,T,S]

  float* out = (float*)d_out;                    // [T,B,E]
  float* avg = out + (size_t)TT * BB * E;        // [B,T,S]

  dim3 blk(256);
  const float scaling = 0.125f;  // HEAD_DIM^-0.5

  convert_kernel<<<(3 * E * E + 255) / 256, blk, 0, stream>>>(ipw, outw, Wh, Owh);

  // Q/K/V projections: M = T*B = 4096, N = E = 1024, K = E = 1024
  gemm_kernel<MODE_Q, false><<<dim3(32, 16, 1), blk, 0, stream>>>(
      query, E, Wh,                     E, ipb,         scaling, 4096, 1024, 1024, qbuf);
  gemm_kernel<MODE_K, false><<<dim3(32, 16, 1), blk, 0, stream>>>(
      key,   E, Wh + (size_t)E * E,     E, ipb + E,     1.0f,    4096, 1024, 1024, kbuf);
  gemm_kernel<MODE_VT, false><<<dim3(32, 16, 1), blk, 0, stream>>>(
      value, E, Wh + (size_t)2 * E * E, E, ipb + 2 * E, 1.0f,    4096, 1024, 1024, vt);

  // scores + masks + softmax -> f16 attention weights
  scores_softmax_kernel<<<BB * NH * (TT / 16), blk, 0, stream>>>(
      qbuf, kbuf, attn_mask, pad, wbuf);

  // head-averaged weights output
  avg_kernel<<<(BB * TT * SS) / 256, blk, 0, stream>>>(wbuf, avg);

  // attn @ V: per (b,h): M = T = 1024, N = D = 64, K = S = 1024
  gemm_kernel<MODE_ATTNOUT, true><<<dim3(8, 1, 64), blk, 0, stream>>>(
      wbuf, SS, vt, SS, nullptr, 1.0f, 1024, 64, 1024, aout);

  // output projection: M = 4096, N = 1024, K = 1024, fp32 out
  gemm_kernel<MODE_FINAL, true><<<dim3(32, 16, 1), blk, 0, stream>>>(
      aout, E, Owh, E, outb, 1.0f, 4096, 1024, 1024, out);
}
